// CompGCN_70025146794712
// MI455X (gfx1250) — matrix operand order repaired and checked by reading, not compiled, not run on hardware
//
#include <hip/hip_runtime.h>
#include <hip/hip_bf16.h>

// ---------------- problem constants (match reference) ----------------
#define N_NODES 32768
#define NE      524288
#define HALF_E  262144
#define DIN     128
#define DH      256
#define NREL    200
#define NG      256
#define NPG     128
#define BN_EPS  1e-5f

typedef __attribute__((ext_vector_type(16))) __bf16 v16bf;
typedef __attribute__((ext_vector_type(8)))  float  v8f;

// f32 -> bf16 (native cast: lets clang use hw cvt/pack ops)
__device__ __forceinline__ __bf16 f2bf(float x) { return (__bf16)x; }

// ---- A fragment: 16x32 bf16, row-major f32 source, optional per-K scale ----
// lane m (m = lane&15) holds row m; g = lane>>4 selects K halves:
//   halfs 0..7  -> K = kb + g*8 + i
//   halfs 8..15 -> K = kb + 16 + g*8 + i
template <bool HAS_SCALE>
__device__ __forceinline__ v16bf load_a(const float* __restrict__ arow, int kb, int g,
                                        const float* __restrict__ scale) {
  v16bf a;
  int k1 = kb + g * 8;
  int k2 = kb + 16 + g * 8;
#pragma unroll
  for (int i = 0; i < 8; ++i) {
    float v1 = arow[k1 + i];
    float v2 = arow[k2 + i];
    if (HAS_SCALE) { v1 *= scale[k1 + i]; v2 *= scale[k2 + i]; }
    a[i]     = f2bf(v1);
    a[8 + i] = f2bf(v2);
  }
  return a;
}

// ---- B fragment from pre-swizzled bf16 weights ----
// Swizzled layout: Wsw[((kbIdx*2 + g)*DH + col)*16 + i], i = half index of the
// fragment (k = kb + g*8 + (i&7) + 16*(i>>3)).  One 32-byte vector load per frag.
__device__ __forceinline__ v16bf load_b_sw(const __bf16* __restrict__ Wsw,
                                           int kbIdx, int g, int col) {
  const v16bf* p = (const v16bf*)(Wsw + ((((size_t)kbIdx * 2 + g) * DH + col) << 4));
  return *p;
}

// pre-swizzle W[K][DH] (f32) -> fragment-ready bf16
__global__ void swizzle_w(const float* __restrict__ W, __bf16* __restrict__ Wsw, int K) {
  int idx = blockIdx.x * blockDim.x + threadIdx.x;   // K*DH elements total
  int i     = idx & 15;
  int col   = (idx >> 4) & (DH - 1);
  int g     = (idx >> 12) & 1;
  int kbIdx = idx >> 13;
  int k = kbIdx * 32 + g * 8 + (i & 7) + ((i >> 3) << 4);
  if (k < K) Wsw[idx] = f2bf(W[(size_t)k * DH + col]);
}

// accumulate A(16xK) @ W(Kx256) restricted to 64-col slab starting at nBase
template <int K, bool HAS_SCALE>
__device__ __forceinline__ void gemm_pass(const float* __restrict__ A,
                                          const __bf16* __restrict__ Wsw,
                                          const float* __restrict__ scale,
                                          int row, int g, int colB, int nBase,
                                          v8f acc[4]) {
  const float* arow = A + (size_t)row * K;
#pragma unroll
  for (int kb = 0; kb < K; kb += 32) {
    v16bf a = load_a<HAS_SCALE>(arow, kb, g, scale);
    int kbIdx = kb >> 5;
#pragma unroll
    for (int j = 0; j < 4; ++j) {
      v16bf b = load_b_sw(Wsw, kbIdx, g, nBase + j * 16 + colB);
      acc[j] = __builtin_amdgcn_wmma_f32_16x16x32_bf16(
          false, a, false, b, (short)0, acc[j], false, false);
    }
  }
}

// ---------------- fused conv GEMM: O = (aggIn@Win + aggOut@Wout + (X*loopRel)@Wloop)/3 + bias
template <int K>
__global__ __launch_bounds__(256) void conv_gemm(
    const float* __restrict__ aggIn, const float* __restrict__ aggOut,
    const float* __restrict__ X, const float* __restrict__ loopRel,
    const __bf16* __restrict__ WswIn, const __bf16* __restrict__ WswOut,
    const __bf16* __restrict__ WswLoop, const float* __restrict__ bias,
    float* __restrict__ O) {
  int lane = threadIdx.x & 31;
  int wave = threadIdx.x >> 5;
  int gw   = blockIdx.x * 8 + wave;      // 8192 waves total
  int tileM = gw >> 2;                   // 0..2047
  int tileN = gw & 3;                    // 0..3
  int g = lane >> 4;
  int mBase = tileM * 16;
  int nBase = tileN * 64;
  int row  = mBase + (lane & 15);
  int colB = lane & 15;

  v8f acc[4] = { v8f{}, v8f{}, v8f{}, v8f{} };

  gemm_pass<K, false>(aggIn,  WswIn,   nullptr, row, g, colB, nBase, acc);
  gemm_pass<K, false>(aggOut, WswOut,  nullptr, row, g, colB, nBase, acc);
  gemm_pass<K, true >(X,      WswLoop, loopRel, row, g, colB, nBase, acc);

#pragma unroll
  for (int j = 0; j < 4; ++j) {
    int col = nBase + j * 16 + (lane & 15);
    float bv = bias[col];
#pragma unroll
    for (int i = 0; i < 8; ++i) {
      int r = mBase + g * 8 + i;   // C/D layout: VGPR i, lanes 0-15 -> M=i, 16-31 -> M=8+i
      O[(size_t)r * DH + col] = acc[j][i] * (1.0f / 3.0f) + bv;
    }
  }
}

// ---------------- generic guarded GEMM: O[M x 256] = A[M x K] @ Wsw
template <int K>
__global__ __launch_bounds__(256) void rel_gemm(
    const float* __restrict__ A, const __bf16* __restrict__ Wsw,
    float* __restrict__ O, int M) {
  int lane = threadIdx.x & 31;
  int wave = threadIdx.x >> 5;
  int gw   = blockIdx.x * 8 + wave;
  int mTiles = (M + 15) >> 4;
  if (gw >= mTiles * 4) return;
  int tileM = gw >> 2, tileN = gw & 3;
  int g = lane >> 4;
  int mBase = tileM * 16;
  int nBase = tileN * 64;
  int row  = mBase + (lane & 15);
  int colB = lane & 15;

  v8f acc[4] = { v8f{}, v8f{}, v8f{}, v8f{} };
#pragma unroll
  for (int kb = 0; kb < K; kb += 32) {
    v16bf a;
    if (row < M) a = load_a<false>(A + (size_t)row * K, kb, g, nullptr);
    else         { v16bf z = {}; a = z; }
    int kbIdx = kb >> 5;
#pragma unroll
    for (int j = 0; j < 4; ++j) {
      v16bf b = load_b_sw(Wsw, kbIdx, g, nBase + j * 16 + colB);
      acc[j] = __builtin_amdgcn_wmma_f32_16x16x32_bf16(
          false, a, false, b, (short)0, acc[j], false, false);
    }
  }
#pragma unroll
  for (int j = 0; j < 4; ++j) {
    int col = nBase + j * 16 + (lane & 15);
#pragma unroll
    for (int i = 0; i < 8; ++i) {
      int r = mBase + g * 8 + i;
      if (r < M) O[(size_t)r * DH + col] = acc[j][i];
    }
  }
}

// ---------------- edge message scatter: agg[dst] += x[src]*rel[et]*dinv[src]*dinv[dst]
__global__ void scatter_msgs(const int* __restrict__ srcArr, const int* __restrict__ dstArr,
                             const int* __restrict__ etype, const float* __restrict__ x,
                             const float* __restrict__ rel, const float* __restrict__ dinv,
                             float* __restrict__ agg, int K) {
  int e = blockIdx.x;
  int d = threadIdx.x;
  int s = srcArr[e], t = dstArr[e];
  int rt = etype[e];
  float nm = dinv[s] * dinv[t];
  float msg = x[(size_t)s * K + d] * rel[(size_t)rt * K + d] * nm;
  atomicAdd(&agg[(size_t)t * K + d], msg);
}

// ---------------- degree / dinv (edge-structure constants) ----------------
__global__ void count_deg(const int* __restrict__ srcAll,
                          float* __restrict__ degIn, float* __restrict__ degOut) {
  int e = blockIdx.x * blockDim.x + threadIdx.x;
  if (e < HALF_E)      atomicAdd(&degIn[srcAll[e]], 1.0f);
  else if (e < NE)     atomicAdd(&degOut[srcAll[e]], 1.0f);
}

__global__ void make_dinv(const float* __restrict__ deg, float* __restrict__ dinv) {
  int n = blockIdx.x * blockDim.x + threadIdx.x;
  if (n < N_NODES) {
    float d = deg[n];
    dinv[n] = d > 0.f ? 1.0f / sqrtf(d) : 0.f;
  }
}

__global__ void zero_f32(float* __restrict__ p, size_t n) {
  size_t i = (size_t)blockIdx.x * blockDim.x + threadIdx.x;
  size_t stride = (size_t)gridDim.x * blockDim.x;
  for (; i < n; i += stride) p[i] = 0.f;
}

// ---------------- batchnorm: column stats, finalize, apply+relu ----------------
__global__ __launch_bounds__(256) void col_stats(const float* __restrict__ O,
                                                 float* __restrict__ sum,
                                                 float* __restrict__ ssq) {
  int c  = threadIdx.x;
  int r0 = blockIdx.x * 128;
  float s = 0.f, q = 0.f;
  for (int r = 0; r < 128; ++r) {
    float v = O[(size_t)(r0 + r) * DH + c];
    s += v; q += v * v;
  }
  atomicAdd(&sum[c], s);
  atomicAdd(&ssq[c], q);
}

__global__ void bn_finalize(const float* __restrict__ sum, const float* __restrict__ ssq,
                            float* __restrict__ mean, float* __restrict__ inv) {
  int c = threadIdx.x;
  float m   = sum[c] / (float)N_NODES;
  float var = ssq[c] / (float)N_NODES - m * m;
  mean[c] = m;
  inv[c]  = rsqrtf(var + BN_EPS);
}

__global__ void bn_relu(const float* __restrict__ O, const float* __restrict__ mean,
                        const float* __restrict__ inv, const float* __restrict__ gamma,
                        const float* __restrict__ beta, float* __restrict__ xo) {
  size_t i = (size_t)blockIdx.x * blockDim.x + threadIdx.x;
  int c = (int)(i & (DH - 1));
  float v = (O[i] - mean[c]) * inv[c] * gamma[c] + beta[c];
  xo[i] = fmaxf(v, 0.f);
}

// ---------------- head: per-graph mean pool, gather, 1024->2 linear ----------------
__global__ __launch_bounds__(256) void pool_mean(const float* __restrict__ x,
                                                 float* __restrict__ pooled) {
  int gph = blockIdx.x, d = threadIdx.x;
  const float* base = x + (size_t)gph * NPG * DH + d;
  float s = 0.f;
  for (int r = 0; r < NPG; ++r) s += base[(size_t)r * DH];
  pooled[gph * DH + d] = s * (1.0f / (float)NPG);
}

__global__ __launch_bounds__(256) void head(
    const float* __restrict__ pooled, const float* __restrict__ rfin,
    const float* __restrict__ xfin, const int* __restrict__ relLabels,
    const float* __restrict__ linW, const float* __restrict__ linB,
    float* __restrict__ out) {
  __shared__ float red0[256];
  __shared__ float red1[256];
  int gph = blockIdx.x, t = threadIdx.x;
  int rl = relLabels[gph];
  float a0 = 0.f, a1 = 0.f;
#pragma unroll
  for (int seg = 0; seg < 4; ++seg) {
    int i = seg * DH + t;
    float f;
    if (seg == 0)      f = pooled[gph * DH + t];
    else if (seg == 1) f = rfin[(size_t)rl * DH + t];
    else if (seg == 2) f = xfin[(size_t)(gph * NPG) * DH + t];
    else               f = xfin[(size_t)(gph * NPG + 1) * DH + t];
    a0 += f * linW[i * 2 + 0];
    a1 += f * linW[i * 2 + 1];
  }
  red0[t] = a0; red1[t] = a1;
  __syncthreads();
  for (int s = 128; s > 0; s >>= 1) {
    if (t < s) { red0[t] += red0[t + s]; red1[t] += red1[t + s]; }
    __syncthreads();
  }
  if (t == 0) {
    out[gph * 2 + 0] = red0[0] + linB[0];
    out[gph * 2 + 1] = red1[0] + linB[1];
  }
}

// =====================================================================
extern "C" void kernel_launch(void* const* d_in, const int* in_sizes, int n_in,
                              void* d_out, int out_size, void* d_ws, size_t ws_size,
                              hipStream_t stream) {
  (void)in_sizes; (void)n_in; (void)out_size; (void)ws_size;

  const float* x0       = (const float*)d_in[0];
  const float* rel_emb  = (const float*)d_in[1];
  const float* w_in1    = (const float*)d_in[2];
  const float* w_out1   = (const float*)d_in[3];
  const float* w_loop1  = (const float*)d_in[4];
  const float* w_rel1   = (const float*)d_in[5];
  const float* loop_rel1= (const float*)d_in[6];
  const float* bias1    = (const float*)d_in[7];
  const float* gamma1   = (const float*)d_in[8];
  const float* beta1    = (const float*)d_in[9];
  const float* W_in     = (const float*)d_in[10];
  const float* W_out    = (const float*)d_in[11];
  const float* W_loop   = (const float*)d_in[12];
  const float* W_rel    = (const float*)d_in[13];
  const float* Loop_rel = (const float*)d_in[14];
  const float* Bias     = (const float*)d_in[15];
  const float* Gamma    = (const float*)d_in[16];
  const float* Beta     = (const float*)d_in[17];
  const float* lin_w    = (const float*)d_in[18];
  const float* lin_b    = (const float*)d_in[19];
  const int*   edge_index = (const int*)d_in[20];
  const int*   edge_type  = (const int*)d_in[21];
  const int*   rel_labels = (const int*)d_in[23];

  // -------- workspace layout --------
  char* ws = (char*)d_ws;
  size_t off = 0;
  auto alloc = [&](size_t bytes) -> void* {
    void* p = (void*)(ws + off);
    off = (off + bytes + 255) & ~(size_t)255;
    return p;
  };
  const size_t big = (size_t)N_NODES * DH * sizeof(float);
  float* B0 = (float*)alloc(big);
  float* B1 = (float*)alloc(big);
  float* B2 = (float*)alloc(big);  // pre-BN output buffer, reused every layer
  float* B3 = (float*)alloc(big);
  float* rA = (float*)alloc((size_t)256 * DH * sizeof(float));
  float* rB = (float*)alloc((size_t)256 * DH * sizeof(float));
  float* degIn  = (float*)alloc(N_NODES * sizeof(float));
  float* degOut = (float*)alloc(N_NODES * sizeof(float));
  float* dinvIn = (float*)alloc(N_NODES * sizeof(float));
  float* dinvOut= (float*)alloc(N_NODES * sizeof(float));
  float* colSum = (float*)alloc(DH * sizeof(float));
  float* colSsq = (float*)alloc(DH * sizeof(float));
  float* bnMean = (float*)alloc(DH * sizeof(float));
  float* bnInv  = (float*)alloc(DH * sizeof(float));
  float* pooled = (float*)alloc((size_t)NG * DH * sizeof(float));
  __bf16* wswIn   = (__bf16*)alloc((size_t)DH * DH * sizeof(__bf16));
  __bf16* wswOut  = (__bf16*)alloc((size_t)DH * DH * sizeof(__bf16));
  __bf16* wswLoop = (__bf16*)alloc((size_t)DH * DH * sizeof(__bf16));
  __bf16* wswRel  = (__bf16*)alloc((size_t)DH * DH * sizeof(__bf16));

  // edge pointer views: row0 = src, row1 = dst
  const int* srcIn  = edge_index;
  const int* dstIn  = edge_index + NE;
  const int* srcOut = edge_index + HALF_E;
  const int* dstOut = edge_index + NE + HALF_E;
  const int* etIn   = edge_type;
  const int* etOut  = edge_type + HALF_E;

  // -------- degrees / symmetric-norm factors (structure constants) --------
  zero_f32<<<128, 256, 0, stream>>>(degIn,  N_NODES);
  zero_f32<<<128, 256, 0, stream>>>(degOut, N_NODES);
  count_deg<<<NE / 256, 256, 0, stream>>>(edge_index, degIn, degOut);
  make_dinv<<<N_NODES / 256, 256, 0, stream>>>(degIn,  dinvIn);
  make_dinv<<<N_NODES / 256, 256, 0, stream>>>(degOut, dinvOut);

  const int relBlocks = (((NREL + 15) / 16) * 4 + 7) / 8;  // 13 M-tiles * 4 N-tiles / 8 waves

  auto run_layer = [&](const float* xin, int K, const float* relCur,
                       const float* wIn, const float* wOut, const float* wLoop,
                       const float* loopRel, const float* bias,
                       const float* gamma, const float* beta,
                       float* aggA, float* aggB, float* Obuf, float* xnext) {
    size_t nk = (size_t)N_NODES * K;
    int swBlocks = (K * DH) / 256;
    swizzle_w<<<swBlocks, 256, 0, stream>>>(wIn,   wswIn,   K);
    swizzle_w<<<swBlocks, 256, 0, stream>>>(wOut,  wswOut,  K);
    swizzle_w<<<swBlocks, 256, 0, stream>>>(wLoop, wswLoop, K);
    zero_f32<<<2048, 256, 0, stream>>>(aggA, nk);
    zero_f32<<<2048, 256, 0, stream>>>(aggB, nk);
    zero_f32<<<1, 256, 0, stream>>>(colSum, DH);
    zero_f32<<<1, 256, 0, stream>>>(colSsq, DH);
    scatter_msgs<<<HALF_E, K, 0, stream>>>(srcIn,  dstIn,  etIn,  xin, relCur, dinvIn,  aggA, K);
    scatter_msgs<<<HALF_E, K, 0, stream>>>(srcOut, dstOut, etOut, xin, relCur, dinvOut, aggB, K);
    if (K == DIN)
      conv_gemm<DIN><<<1024, 256, 0, stream>>>(aggA, aggB, xin, loopRel,
                                               wswIn, wswOut, wswLoop, bias, Obuf);
    else
      conv_gemm<DH><<<1024, 256, 0, stream>>>(aggA, aggB, xin, loopRel,
                                              wswIn, wswOut, wswLoop, bias, Obuf);
    col_stats<<<256, 256, 0, stream>>>(Obuf, colSum, colSsq);
    bn_finalize<<<1, 256, 0, stream>>>(colSum, colSsq, bnMean, bnInv);
    bn_relu<<<(N_NODES * DH) / 256, 256, 0, stream>>>(Obuf, bnMean, bnInv, gamma, beta, xnext);
  };

  auto run_rel = [&](const float* relCur, const float* wRel, float* relNext, int K) {
    int swBlocks = (K * DH) / 256;
    swizzle_w<<<swBlocks, 256, 0, stream>>>(wRel, wswRel, K);
    if (K == DIN)
      rel_gemm<DIN><<<relBlocks, 256, 0, stream>>>(relCur, wswRel, relNext, NREL);
    else
      rel_gemm<DH><<<relBlocks, 256, 0, stream>>>(relCur, wswRel, relNext, NREL);
  };

  // -------- layer 1: DIN=128 -> DH=256 --------
  run_layer(x0, DIN, rel_emb, w_in1, w_out1, w_loop1, loop_rel1, bias1, gamma1, beta1,
            B0, B1, B2, B3);
  run_rel(rel_emb, w_rel1, rA, DIN);

  // -------- layers 2..5: DH -> DH --------
  float* xcur = B3;
  float* freeA = B0;
  float* freeB = B1;
  float* rcur = rA;
  float* rnext = rB;
  for (int l = 0; l < 4; ++l) {
    run_layer(xcur, DH, rcur,
              W_in + (size_t)l * DH * DH, W_out + (size_t)l * DH * DH,
              W_loop + (size_t)l * DH * DH,
              Loop_rel + (size_t)l * DH, Bias + (size_t)l * DH,
              Gamma + (size_t)l * DH, Beta + (size_t)l * DH,
              freeA, freeB, B2, /*xnext=*/freeA);
    run_rel(rcur, W_rel + (size_t)l * DH * DH, rnext, DH);
    float* nx = freeA; freeA = xcur; xcur = nx;   // rotate x buffers, freeB reusable
    float* rt = rcur; rcur = rnext; rnext = rt;   // rotate rel buffers
  }

  // -------- head --------
  pool_mean<<<NG, 256, 0, stream>>>(xcur, pooled);
  head<<<NG, 256, 0, stream>>>(pooled, rcur, xcur, rel_labels, lin_w, lin_b, (float*)d_out);
}
